// SepTemporalAttention_51651276702312
// MI455X (gfx1250) — matrix operand order repaired and checked by reading, not compiled, not run on hardware
//
#include <hip/hip_runtime.h>
#include <hip/hip_bf16.h>

// ---------------------------------------------------------------------------
// Separable temporal attention for MI455X (gfx1250), bf16 WMMA pipeline.
// Async global->LDS weight staging (ASYNCcnt path) + v_wmma_f32_16x16x32_bf16.
// ---------------------------------------------------------------------------

typedef __attribute__((ext_vector_type(16))) __bf16 v16bf;
typedef __attribute__((ext_vector_type(8)))  float  v8f;

#define NJ   24
#define NB   32
#define NS   120
#define ND   128
#define NH   8
#define DEP  16
#define SP   128      // padded sequence
#define NREL 99       // 2*49+1
#define RPAD 112      // padded rel rows (7 tiles of 16)

union FragB {
    v16bf v;
    unsigned u[8];
    uint4 q[2];
};

__device__ __forceinline__ unsigned short f2bf(float f) {
    unsigned x = __float_as_uint(f);
    x += 0x7fffu + ((x >> 16) & 1u);   // round-to-nearest-even
    return (unsigned short)(x >> 16);
}
__device__ __forceinline__ unsigned pkbf(float a, float b) {
    return (unsigned)f2bf(a) | ((unsigned)f2bf(b) << 16);
}
__device__ __forceinline__ v8f wmma_bf16(v16bf a, v16bf b, v8f c) {
    return __builtin_amdgcn_wmma_f32_16x16x32_bf16(false, a, false, b,
                                                   (short)0, c, false, false);
}

// Async DMA one 16-byte chunk global -> LDS (no VGPR round trip, ASYNCcnt).
// VDST = single VGPR with 32-bit LDS byte address (low 32 bits of the
// LDS-aperture generic pointer), SADDR = uniform 64-bit base, VADDR = offset.
__device__ __forceinline__ void async_copy_b128(unsigned short* lds_dst,
                                                const unsigned short* g_base,
                                                unsigned byte_off) {
    unsigned lds_addr = (unsigned)(unsigned long long)lds_dst;
    asm volatile("global_load_async_to_lds_b128 %0, %1, %2 offset:0"
                 :
                 : "v"(lds_addr), "v"(byte_off), "s"(g_base)
                 : "memory");
}
__device__ __forceinline__ void wait_async0() {
    asm volatile("s_wait_asynccnt 0" ::: "memory");
}

// ---------------------------------------------------------------------------
// Kernel 1: prep — bf16 weight conversion + positional encoding + rel tables
// ---------------------------------------------------------------------------
__global__ void prep_kernel(const float* __restrict__ Wq, const float* __restrict__ Wk,
                            const float* __restrict__ Wv, const float* __restrict__ Wo,
                            const float* __restrict__ keyt, const float* __restrict__ valt,
                            unsigned short* wqb, unsigned short* wkb, unsigned short* wvb,
                            unsigned short* wob, unsigned short* ktb, unsigned short* vttb,
                            float* pe) {
    int tid = blockIdx.x * blockDim.x + threadIdx.x;
    int stride = gridDim.x * blockDim.x;
    const int NW = NJ * ND * ND;
    for (int i = tid; i < NW; i += stride) {
        wqb[i] = f2bf(Wq[i]);
        wkb[i] = f2bf(Wk[i]);
        wvb[i] = f2bf(Wv[i]);
    }
    for (int i = tid; i < ND * ND; i += stride) wob[i] = f2bf(Wo[i]);
    // key table, padded to RPAD rows of DEP (row-major)
    for (int i = tid; i < RPAD * DEP; i += stride) {
        int r = i / DEP;
        ktb[i] = (r < NREL) ? f2bf(keyt[i]) : (unsigned short)0;
    }
    // value table transposed: [DEP][128], zero-padded cols >= NREL
    for (int i = tid; i < DEP * 128; i += stride) {
        int d = i / 128, r = i % 128;
        vttb[i] = (r < NREL) ? f2bf(valt[r * DEP + d]) : (unsigned short)0;
    }
    // positional encoding [NS][ND]
    for (int i = tid; i < NS * ND; i += stride) {
        int s = i / ND, c = i % ND;
        float e = 2.0f * (float)(c / 2) / (float)ND;
        float angle = (float)s * __powf(10000.0f, -e);
        pe[i] = (c & 1) ? __cosf(angle) : __sinf(angle);
    }
}

// ---------------------------------------------------------------------------
// Kernel 2: QKV projection. grid=(60, NJ), block=128 (4 waves x 16 rows)
// q,k stored [j][b][h][SP][DEP] bf16 ; v stored transposed [j][b][h][DEP][SP]
// ---------------------------------------------------------------------------
__global__ __launch_bounds__(128) void proj_kernel(
        const float* __restrict__ x, const float* __restrict__ pe,
        const unsigned short* __restrict__ wqb, const unsigned short* __restrict__ wkb,
        const unsigned short* __restrict__ wvb,
        const float* __restrict__ bq, const float* __restrict__ bk,
        const float* __restrict__ bv,
        unsigned short* Qw, unsigned short* Kw, unsigned short* VTw) {
    __shared__ unsigned short sX[64][128];   // 16 KB
    __shared__ unsigned short sW[128][128];  // 32 KB
    const int j = blockIdx.y;
    const int rb = blockIdx.x;
    const int tid = threadIdx.x;
    const int wave = tid >> 5, lane = tid & 31;
    const int n = lane & 15, hi = lane >> 4;

    // stage (x + PE) as bf16 pairs
    for (int it = tid; it < 64 * 64; it += 128) {
        int row = it / 64;
        int cp = (it % 64) * 2;
        int g = rb * 64 + row;
        int b_ = g / NS, s_ = g % NS;
        const float* xr = x + ((size_t)(b_ * NS + s_) * NJ + j) * ND;
        const float* per = pe + s_ * ND;
        *reinterpret_cast<unsigned*>(&sX[row][cp]) =
            pkbf(xr[cp] + per[cp], xr[cp + 1] + per[cp + 1]);
    }

    int gb[8], gs[8];
    for (int r = 0; r < 8; r++) {
        int g = rb * 64 + wave * 16 + r + 8 * hi;
        gb[r] = g / NS;
        gs[r] = g % NS;
    }

    for (int wt = 0; wt < 3; wt++) {
        const unsigned short* wsrc =
            (wt == 0 ? wqb : wt == 1 ? wkb : wvb) + (size_t)j * ND * ND;
        __syncthreads();
        // async DMA the 32KB weight tile straight into LDS (no VGPR staging)
        for (int i = tid; i < 2048; i += 128)
            async_copy_b128(&sW[0][0] + i * 8, wsrc, (unsigned)(i * 16));
        wait_async0();
        __syncthreads();

        v8f acc[8];
        for (int nt = 0; nt < 8; nt++) acc[nt] = (v8f){0,0,0,0,0,0,0,0};
        const unsigned short* xrow = &sX[wave * 16 + n][0];
        for (int ks = 0; ks < 128; ks += 32) {
            FragB A;
            A.q[0] = *reinterpret_cast<const uint4*>(xrow + ks + hi * 8);
            A.q[1] = *reinterpret_cast<const uint4*>(xrow + ks + 16 + hi * 8);
            for (int nt = 0; nt < 8; nt++) {
                const unsigned short* wrow = &sW[nt * 16 + n][0];
                FragB Bf;
                Bf.q[0] = *reinterpret_cast<const uint4*>(wrow + ks + hi * 8);
                Bf.q[1] = *reinterpret_cast<const uint4*>(wrow + ks + 16 + hi * 8);
                acc[nt] = wmma_bf16(A.v, Bf.v, acc[nt]);
            }
        }

        const float* bias = (wt == 0 ? bq : wt == 1 ? bk : bv) + j * ND;
        for (int nt = 0; nt < 8; nt++) {
            float bval = bias[nt * 16 + n];
            for (int r = 0; r < 8; r++) {
                unsigned short v16 = f2bf(acc[nt][r] + bval);
                size_t jbh = ((size_t)j * NB + gb[r]) * NH + nt;
                if (wt == 0)      Qw[(jbh * SP + gs[r]) * DEP + n] = v16;
                else if (wt == 1) Kw[(jbh * SP + gs[r]) * DEP + n] = v16;
                else              VTw[(jbh * DEP + n) * SP + gs[r]] = v16;
            }
        }
    }
}

// ---------------------------------------------------------------------------
// Kernel 3: attention. grid = NJ*NB*2, block = 128 (wave = head)
// ---------------------------------------------------------------------------
__global__ __launch_bounds__(128) void attn_kernel(
        const unsigned short* __restrict__ Qw, const unsigned short* __restrict__ Kw,
        const unsigned short* __restrict__ VTw,
        const unsigned short* __restrict__ ktb, const unsigned short* __restrict__ vttb,
        unsigned short* aout, float* attn_out) {
    __shared__ float sP[4][16][128];           // P / w_rel, per wave (32 KB)
    __shared__ unsigned short sWl[4][16][128]; // bf16 softmax weights  (16 KB)
    const int bi = blockIdx.x;
    const int pair = bi & 1, jb = bi >> 1;
    const int j = jb / NB, b = jb % NB;
    const int tid = threadIdx.x, wave = tid >> 5, lane = tid & 31;
    const int h = pair * 4 + wave;
    const int n = lane & 15, hi = lane >> 4;
    const size_t jbh = ((size_t)j * NB + b) * NH + h;
    const unsigned short* qbase = Qw + jbh * SP * DEP;
    const unsigned short* kbase = Kw + jbh * SP * DEP;
    const unsigned short* vbase = VTw + jbh * DEP * SP;
    float (*P)[128] = sP[wave];
    unsigned short (*W)[128] = sWl[wave];
    const uint4 z4 = {0, 0, 0, 0};

    for (int m = 0; m < 8; m++) {
        __syncthreads();   // previous w_rel reads done before reuse of sP

        // ---- A fragment: q strip rows m*16..m*16+15, K = depth (pad to 32)
        FragB A;
        A.q[0] = *reinterpret_cast<const uint4*>(qbase + (m * 16 + n) * DEP + hi * 8);
        A.q[1] = z4;

        // ---- P = q @ key_table^T   -> LDS  (cols 0..111)
        for (int nt = 0; nt < 7; nt++) {
            FragB Bf;
            {
                const unsigned short* kr = ktb + (nt * 16 + n) * DEP;
                Bf.q[0] = *reinterpret_cast<const uint4*>(kr + hi * 8);
                Bf.q[1] = z4;
            }
            v8f c = (v8f){0,0,0,0,0,0,0,0};
            c = wmma_bf16(A.v, Bf.v, c);
            for (int r = 0; r < 8; r++) P[r + 8 * hi][nt * 16 + n] = c[r];
        }
        __syncthreads();

        // ---- logits = q @ k^T + gather(P), scale, causal mask
        v8f acc[8];
        for (int nt = 0; nt < 8; nt++) {
            FragB Bf;
            {
                const unsigned short* kr = kbase + (nt * 16 + n) * DEP;
                Bf.q[0] = *reinterpret_cast<const uint4*>(kr + hi * 8);
                Bf.q[1] = z4;
            }
            v8f c = (v8f){0,0,0,0,0,0,0,0};
            c = wmma_bf16(A.v, Bf.v, c);
            for (int r = 0; r < 8; r++) {
                int srow = m * 16 + r + 8 * hi;
                int t = nt * 16 + n;
                int dd = t - srow;
                dd = dd < -49 ? -49 : (dd > 49 ? 49 : dd);
                float lg = (c[r] + P[r + 8 * hi][dd + 49]) * 0.25f;
                if (t > srow) lg -= 1e9f;        // look-ahead mask
                if (t >= NS)  lg = -3.0e38f;     // padding
                c[r] = lg;
            }
            acc[nt] = c;
        }

        // ---- fp32 softmax over t (row lives in 16 lanes of one half)
        for (int r = 0; r < 8; r++) {
            float mx = -3.4e38f;
            for (int nt = 0; nt < 8; nt++) mx = fmaxf(mx, acc[nt][r]);
            for (int msk = 1; msk < 16; msk <<= 1)
                mx = fmaxf(mx, __shfl_xor(mx, msk, 32));
            float sm = 0.0f;
            for (int nt = 0; nt < 8; nt++) {
                float e = __expf(acc[nt][r] - mx);
                acc[nt][r] = e;
                sm += e;
            }
            for (int msk = 1; msk < 16; msk <<= 1) sm += __shfl_xor(sm, msk, 32);
            float inv = 1.0f / sm;
            for (int nt = 0; nt < 8; nt++) acc[nt][r] *= inv;
        }

        // ---- attention weights for last query row (s = 119 -> r=7, hi=0)
        if (m == 7 && hi == 0) {
            for (int nt = 0; nt < 8; nt++) {
                int t = nt * 16 + n;
                if (t < NS)
                    attn_out[((size_t)(b * NJ + j) * NH + h) * NS + t] = acc[nt][7];
            }
        }

        // ---- stash w (bf16) and zero w_rel
        for (int nt = 0; nt < 8; nt++)
            for (int r = 0; r < 8; r++)
                W[r + 8 * hi][nt * 16 + n] = f2bf(acc[nt][r]);
        for (int i = lane; i < 2048; i += 32) (&P[0][0])[i] = 0.0f;
        __syncthreads();

        // ---- scatter-add w into w_rel (causal => only lower clip collides)
        for (int nt = 0; nt < 8; nt++)
            for (int r = 0; r < 8; r++) {
                int srow = m * 16 + r + 8 * hi;
                int t = nt * 16 + n;
                if (t <= srow && t < NS) {
                    int dd = t - srow;
                    dd = dd < -49 ? -49 : dd;
                    atomicAdd(&P[r + 8 * hi][dd + 49], acc[nt][r]);
                }
            }
        __syncthreads();

        // ---- out = w @ v   +   w_rel @ val_table
        v8f oc = (v8f){0,0,0,0,0,0,0,0};
        const unsigned short* wrow = &W[n][0];
        for (int ks = 0; ks < 128; ks += 32) {
            FragB Aw;
            Aw.q[0] = *reinterpret_cast<const uint4*>(wrow + ks + hi * 8);
            Aw.q[1] = *reinterpret_cast<const uint4*>(wrow + ks + 16 + hi * 8);
            const unsigned short* vrow = vbase + n * SP;
            FragB Bv;
            Bv.q[0] = *reinterpret_cast<const uint4*>(vrow + ks + hi * 8);
            Bv.q[1] = *reinterpret_cast<const uint4*>(vrow + ks + 16 + hi * 8);
            oc = wmma_bf16(Aw.v, Bv.v, oc);
        }
        const float* wrl = &P[n][0];
        for (int ks = 0; ks < 128; ks += 32) {
            FragB Ar;
            for (int p = 0; p < 8; p++) {
                int K = ks + (p < 4 ? hi * 8 + 2 * p : 16 + hi * 8 + 2 * (p - 4));
                Ar.u[p] = pkbf(wrl[K], wrl[K + 1]);
            }
            const unsigned short* vtr = vttb + n * 128;
            FragB Bt;
            Bt.q[0] = *reinterpret_cast<const uint4*>(vtr + ks + hi * 8);
            Bt.q[1] = *reinterpret_cast<const uint4*>(vtr + ks + 16 + hi * 8);
            oc = wmma_bf16(Ar.v, Bt.v, oc);
        }
        for (int r = 0; r < 8; r++) {
            int srow = m * 16 + r + 8 * hi;
            if (srow < NS)
                aout[(((size_t)j * NB + b) * NS + srow) * ND + h * 16 + n] =
                    f2bf(oc[r]);
        }
    }
}

// ---------------------------------------------------------------------------
// Kernel 4: output projection. grid = 1440, block = 128
// ---------------------------------------------------------------------------
__global__ __launch_bounds__(128) void out_kernel(
        const unsigned short* __restrict__ aout,
        const unsigned short* __restrict__ wob,
        const float* __restrict__ bo, float* out) {
    __shared__ unsigned short sW[128][128];
    const int rb = blockIdx.x;
    const int tid = threadIdx.x, wave = tid >> 5, lane = tid & 31;
    const int n = lane & 15, hi = lane >> 4;
    // async DMA Wo tile into LDS
    for (int i = tid; i < 2048; i += 128)
        async_copy_b128(&sW[0][0] + i * 8, wob, (unsigned)(i * 16));
    wait_async0();
    __syncthreads();

    const unsigned short* arow = aout + ((size_t)rb * 64 + wave * 16 + n) * ND;
    v8f acc[8];
    for (int nt = 0; nt < 8; nt++) acc[nt] = (v8f){0,0,0,0,0,0,0,0};
    for (int ks = 0; ks < 128; ks += 32) {
        FragB A;
        A.q[0] = *reinterpret_cast<const uint4*>(arow + ks + hi * 8);
        A.q[1] = *reinterpret_cast<const uint4*>(arow + ks + 16 + hi * 8);
        for (int nt = 0; nt < 8; nt++) {
            const unsigned short* wrow = &sW[nt * 16 + n][0];
            FragB Bf;
            Bf.q[0] = *reinterpret_cast<const uint4*>(wrow + ks + hi * 8);
            Bf.q[1] = *reinterpret_cast<const uint4*>(wrow + ks + 16 + hi * 8);
            acc[nt] = wmma_bf16(A.v, Bf.v, acc[nt]);
        }
    }
    for (int nt = 0; nt < 8; nt++) {
        int e = nt * 16 + n;
        float bval = bo[e];
        for (int r = 0; r < 8; r++) {
            int g = rb * 64 + wave * 16 + r + 8 * hi;
            int j = g / (NB * NS);
            int rem = g - j * (NB * NS);
            int b_ = rem / NS, s_ = rem % NS;
            out[((size_t)(b_ * NS + s_) * NJ + j) * ND + e] = acc[nt][r] + bval;
        }
    }
}

// ---------------------------------------------------------------------------
// Host launcher
// ---------------------------------------------------------------------------
static size_t align256(size_t x) { return (x + 255) & ~(size_t)255; }

extern "C" void kernel_launch(void* const* d_in, const int* in_sizes, int n_in,
                              void* d_out, int out_size, void* d_ws, size_t ws_size,
                              hipStream_t stream) {
    const float* x    = (const float*)d_in[0];
    // d_in[1] = mask (implicit causal; recomputed on device)
    const float* Wq   = (const float*)d_in[2];
    const float* bq   = (const float*)d_in[3];
    const float* Wk   = (const float*)d_in[4];
    const float* bk   = (const float*)d_in[5];
    const float* Wv   = (const float*)d_in[6];
    const float* bv   = (const float*)d_in[7];
    const float* Wo   = (const float*)d_in[8];
    const float* bo   = (const float*)d_in[9];
    const float* keyt = (const float*)d_in[10];
    const float* valt = (const float*)d_in[11];
    float* out = (float*)d_out;
    float* attn_out = out + (size_t)NB * NS * NJ * ND;

    char* ws = (char*)d_ws;
    size_t off = 0;
    auto take = [&](size_t bytes) { char* p = ws + off; off = align256(off + bytes); return p; };
    unsigned short* wqb  = (unsigned short*)take((size_t)NJ * ND * ND * 2);
    unsigned short* wkb  = (unsigned short*)take((size_t)NJ * ND * ND * 2);
    unsigned short* wvb  = (unsigned short*)take((size_t)NJ * ND * ND * 2);
    unsigned short* wob  = (unsigned short*)take((size_t)ND * ND * 2);
    unsigned short* ktb  = (unsigned short*)take((size_t)RPAD * DEP * 2);
    unsigned short* vttb = (unsigned short*)take((size_t)DEP * 128 * 2);
    float*          pe   = (float*)take((size_t)NS * ND * 4);
    unsigned short* Qw   = (unsigned short*)take((size_t)NJ * NB * NH * SP * DEP * 2);
    unsigned short* Kw   = (unsigned short*)take((size_t)NJ * NB * NH * SP * DEP * 2);
    unsigned short* VTw  = (unsigned short*)take((size_t)NJ * NB * NH * DEP * SP * 2);
    unsigned short* aout = (unsigned short*)take((size_t)NJ * NB * NS * ND * 2);
    (void)ws_size; (void)in_sizes; (void)n_in; (void)out_size;

    prep_kernel<<<128, 256, 0, stream>>>(Wq, Wk, Wv, Wo, keyt, valt,
                                         wqb, wkb, wvb, wob, ktb, vttb, pe);

    dim3 pg((NB * NS) / 64, NJ);
    proj_kernel<<<pg, 128, 0, stream>>>(x, pe, wqb, wkb, wvb, bq, bk, bv,
                                        Qw, Kw, VTw);

    attn_kernel<<<NJ * NB * 2, 128, 0, stream>>>(Qw, Kw, VTw, ktb, vttb,
                                                 aout, attn_out);

    out_kernel<<<(NJ * NB * NS) / 64, 128, 0, stream>>>(aout, wob, bo, out);
}
